// TERMAttention_34445637714729
// MI455X (gfx1250) — matrix operand order repaired and checked by reading, not compiled, not run on hardware
//
#include <hip/hip_runtime.h>

typedef __attribute__((ext_vector_type(16))) __bf16 v16bf;
typedef __attribute__((ext_vector_type(8)))  float  v8f;
typedef __attribute__((ext_vector_type(4)))  unsigned int v4u;
typedef __attribute__((ext_vector_type(8)))  int v8i;
typedef __attribute__((ext_vector_type(4)))  int v4i;

#define LDW 136   // padded row stride for 128-wide arrays (272B = multiple of 16B)
#define LDV 40    // padded row stride for 32-wide arrays (80B)
#define NEG_INF   (-3.402823466e38f)
#define INV_SQRT_D 0.17677669529663688f   // 1/sqrt(32)

#define W_USHORTS (2 * 128 * LDW)          // one weight image (hi+lo) in ushorts = 34816
#define W_BYTES   (W_USHORTS * 2)          // 69632 bytes
#define OUT_TILE_BYTES (32 * 128 * 4)      // 16384 bytes

#if __has_builtin(__builtin_amdgcn_tensor_load_to_lds) && \
    __has_builtin(__builtin_amdgcn_tensor_store_from_lds) && \
    __has_builtin(__builtin_amdgcn_s_wait_tensorcnt)
#define HAVE_TDM 1
#else
#define HAVE_TDM 0
#endif

union Frag16 { v16bf v; uint4 q[2]; };

__device__ __forceinline__ unsigned short f2bf(float f) {
  unsigned u = __float_as_uint(f);
  u += 0x7FFFu + ((u >> 16) & 1u);      // round-to-nearest-even
  return (unsigned short)(u >> 16);
}
__device__ __forceinline__ float bf2f(unsigned short h) {
  return __uint_as_float(((unsigned)h) << 16);
}

// A operand: 16x32 (MxK). Lane l (r=l&15, hh=l>>4) holds row m0+r,
// K = k0 + 8*hh + {0..7} and k0 + 16 + 8*hh + {0..7}  (ISA 7.12.2).
__device__ __forceinline__ v16bf loadA(const unsigned short* base, int ld,
                                       int m0, int k0, int lane) {
  const unsigned short* p = base + (m0 + (lane & 15)) * ld + k0 + 8 * (lane >> 4);
  Frag16 u;
  u.q[0] = *(const uint4*)(p);
  u.q[1] = *(const uint4*)(p + 16);
  return u.v;
}
// B operand: 32x16 (KxN). Lane l holds column n0+(l&15),
// K = k0 + 8*(l>>4) + {0..7} and k0 + 16 + 8*(l>>4) + {0..7}. base is stored [n][k] (contiguous k).
__device__ __forceinline__ v16bf loadB(const unsigned short* base, int ld,
                                       int n0, int k0, int lane) {
  const unsigned short* p = base + (n0 + (lane & 15)) * ld + k0 + 8 * (lane >> 4);
  Frag16 u;
  u.q[0] = *(const uint4*)(p);
  u.q[1] = *(const uint4*)(p + 16);
  return u.v;
}

__device__ __forceinline__ v8f wmma_bf16(v16bf a, v16bf b, v8f c) {
  return __builtin_amdgcn_wmma_f32_16x16x32_bf16(false, a, false, b,
                                                 (short)0, c, false, false);
}
// split-precision product: (ah+al)*(bh+bl) ~= ah*bh + ah*bl + al*bh
__device__ __forceinline__ v8f mm3(v16bf ah, v16bf al, v16bf bh, v16bf bl, v8f c) {
  c = wmma_bf16(ah, bh, c);
  c = wmma_bf16(ah, bl, c);
  c = wmma_bf16(al, bh, c);
  return c;
}

__device__ __forceinline__ unsigned lds_offset(const void* p) {
  // generic pointer to LDS: addr[31:0] is the LDS byte offset (aperture rule)
  return (unsigned)(unsigned long long)(uintptr_t)p;
}

#if HAVE_TDM
// 1-D TDM copy descriptor: n8 elements of 8 bytes, contiguous both sides.
// D# bitfields per CDNA5 ISA 8.3/8.4: group0 = {count|flags, lds_addr,
// global_addr[31:0], global_addr[56:32] | type=2<<30}; group1 packs
// data_size=8B, tensor_dim0=n8, tensor_dim1=1, tile_dim0=n8, tile_dim1=1.
__device__ __forceinline__ void tdm_desc(v4u& g0, v8i& g1,
                                         unsigned lds, unsigned long long ga,
                                         unsigned n8) {
  g0[0] = 1u;                                         // count=1 (valid descriptor)
  g0[1] = lds;                                        // lds_addr (bytes)
  g0[2] = (unsigned)(ga & 0xFFFFFFFFull);             // global_addr[31:0]
  g0[3] = (unsigned)((ga >> 32) & 0x1FFFFFFull)       // global_addr[56:32]
        | (2u << 30);                                 // type=2 ("image")
  g1[0] = (int)(3u << 16);                            // data_size=3 (8 bytes)
  g1[1] = (int)((n8 & 0xFFFFu) << 16);                // tensor_dim0[15:0]
  g1[2] = (int)(((n8 >> 16) & 0xFFFFu) | (1u << 16)); // tensor_dim0[31:16], tensor_dim1=1
  g1[3] = (int)((n8 & 0xFFFFu) << 16);                // tile_dim0
  g1[4] = 1;                                          // tile_dim1=1, tile_dim2=0
  g1[5] = (int)n8;                                    // tensor_dim0_stride[31:0]
  g1[6] = 0;
  g1[7] = 0;
}
__device__ __forceinline__ void tdm_load_1d(unsigned lds, unsigned long long ga,
                                            unsigned n8) {
  v4u g0; v8i g1; v4i z4 = {0, 0, 0, 0};
  v8i z8 = {0, 0, 0, 0, 0, 0, 0, 0};
  tdm_desc(g0, g1, lds, ga, n8);
  __builtin_amdgcn_tensor_load_to_lds(g0, g1, z4, z4, z8, 0);
}
__device__ __forceinline__ void tdm_store_1d(unsigned lds, unsigned long long ga,
                                             unsigned n8) {
  v4u g0; v8i g1; v4i z4 = {0, 0, 0, 0};
  v8i z8 = {0, 0, 0, 0, 0, 0, 0, 0};
  tdm_desc(g0, g1, lds, ga, n8);
  __builtin_amdgcn_tensor_store_from_lds(g0, g1, z4, z4, z8, 0);
}
#endif

// ---- prep kernel: convert W[k][n] fp32 -> transposed padded split-bf16 image
// ws layout per weight: hi[128*LDW] then lo[128*LDW]  (exact LDS image)
__global__ __launch_bounds__(256)
void prep_weights(const float* __restrict__ Wq, const float* __restrict__ Wk,
                  const float* __restrict__ Wv, const float* __restrict__ Wo,
                  unsigned short* __restrict__ ws) {
  const float* W = (blockIdx.x == 0) ? Wq : (blockIdx.x == 1) ? Wk
                 : (blockIdx.x == 2) ? Wv : Wo;
  unsigned short* hi = ws + (size_t)blockIdx.x * W_USHORTS;
  unsigned short* lo = hi + 128 * LDW;
  for (int idx = threadIdx.x; idx < 128 * LDW; idx += 256) {
    int n = idx / LDW, k = idx - n * LDW;
    float v = (k < 128) ? W[k * 128 + n] : 0.0f;
    unsigned short h = f2bf(v);
    hi[idx] = h;
    lo[idx] = f2bf(v - bf2f(h));
  }
}

template <int USE_WS>
__global__ __launch_bounds__(256)
void term_attn_kernel(const float* __restrict__ src, const int* __restrict__ mask,
                      const float* __restrict__ Wq, const float* __restrict__ Wk,
                      const float* __restrict__ Wv, const float* __restrict__ Wo,
                      const unsigned short* __restrict__ wsbf,
                      float* __restrict__ out) {
  __shared__ __align__(16) unsigned short sW[2][128 * LDW];  // weights, transposed [n][k]
  __shared__ __align__(16) unsigned short sS[2][32 * LDW];   // src tile; reused as O
  __shared__ __align__(16) unsigned short sQ[2][32 * LDW];   // Q; reused as f32 out staging
  __shared__ __align__(16) unsigned short sK[2][32 * LDW];
  __shared__ __align__(16) unsigned short sVt[2][128 * LDV]; // V transposed [d][c]
  __shared__ int sMask[32];

  // During attention the W buffer is dead: alias logits (f32) and probs onto it.
  float* sLog = (float*)(&sW[0][0]);            // [4][32][32] f32 = 16KB in sW[0]
  unsigned short* sPhi = &sW[1][0];             // probs hi: [a][128] (head h at cols 32h..)
  unsigned short* sPlo = &sW[1][32 * LDW];      // probs lo

  const int tid  = threadIdx.x;
  const int lane = tid & 31;
  const int wave = tid >> 5;
  const int nl   = lane & 15;
  const int hh   = lane >> 4;
  const size_t tile = blockIdx.x;

  const float* Sg = src + tile * (32 * 128);
  float*       Og = out + tile * (32 * 128);

  // ---- load src tile 32x128 fp32 (vectorized) -> bf16 hi/lo in LDS ----
  {
    const float4* S4 = (const float4*)Sg;
    #pragma unroll
    for (int i = 0; i < 4; ++i) {
      int idx4 = tid * 4 + i;                 // 0..1023, 16B-contiguous per thread
      float4 v4 = S4[idx4];
      int idx = idx4 * 4;
      int r = idx >> 7, c = idx & 127;
      float vv[4] = {v4.x, v4.y, v4.z, v4.w};
      #pragma unroll
      for (int j = 0; j < 4; ++j) {
        unsigned short h = f2bf(vv[j]);
        sS[0][r * LDW + c + j] = h;
        sS[1][r * LDW + c + j] = f2bf(vv[j] - bf2f(h));
      }
    }
  }
  if (tid < 32) sMask[tid] = mask[tile * 32 + tid];

  // bring weight image `widx` into sW (call with all threads; barrier follows at caller)
  auto loadW = [&](int widx, const float* W) {
    if (USE_WS) {
#if HAVE_TDM
      if (wave == 0) {
        tdm_load_1d(lds_offset(&sW[0][0]),
                    (unsigned long long)(uintptr_t)(wsbf + (size_t)widx * W_USHORTS),
                    W_BYTES / 8);
        __builtin_amdgcn_s_wait_tensorcnt(0);
      }
#else
      const uint4* s = (const uint4*)(wsbf + (size_t)widx * W_USHORTS);
      uint4* d = (uint4*)&sW[0][0];
      #pragma unroll
      for (int i = 0; i < W_USHORTS / 8 / 256; ++i) d[tid + i * 256] = s[tid + i * 256];
#endif
    } else {
      #pragma unroll
      for (int i = 0; i < 64; ++i) {
        int idx = tid + i * 256;               // 0..16383
        int k = idx >> 7, n = idx & 127;
        float v = W[idx];
        unsigned short h = f2bf(v);
        sW[0][n * LDW + k] = h;
        sW[1][n * LDW + k] = f2bf(v - bf2f(h));
      }
    }
  };

  // 32x128 = sS(32x128) @ W(128x128); each wave computes cols [16w,16w+16)
  // mode 0: store split bf16 row-major [m][n]; mode 1: transposed [n][m] stride LDV
  auto project = [&](unsigned short* Dhi, unsigned short* Dlo, int mode) {
    const int n0 = wave * 16;
    v8f acc0 = {0.f,0.f,0.f,0.f,0.f,0.f,0.f,0.f};
    v8f acc1 = {0.f,0.f,0.f,0.f,0.f,0.f,0.f,0.f};
    #pragma unroll
    for (int kb = 0; kb < 4; ++kb) {
      int k0 = kb * 32;
      v16bf bh = loadB(sW[0], LDW, n0, k0, lane);
      v16bf bl = loadB(sW[1], LDW, n0, k0, lane);
      v16bf ah = loadA(sS[0], LDW, 0, k0, lane);
      v16bf al = loadA(sS[1], LDW, 0, k0, lane);
      acc0 = mm3(ah, al, bh, bl, acc0);
      ah = loadA(sS[0], LDW, 16, k0, lane);
      al = loadA(sS[1], LDW, 16, k0, lane);
      acc1 = mm3(ah, al, bh, bl, acc1);
    }
    #pragma unroll
    for (int mt = 0; mt < 2; ++mt) {
      v8f a = mt ? acc1 : acc0;
      #pragma unroll
      for (int r = 0; r < 8; ++r) {
        int m = mt * 16 + r + 8 * hh;
        float v = a[r];
        unsigned short h = f2bf(v);
        unsigned short l = f2bf(v - bf2f(h));
        if (mode == 1) { Dhi[(n0 + nl) * LDV + m] = h; Dlo[(n0 + nl) * LDV + m] = l; }
        else           { Dhi[m * LDW + n0 + nl] = h;  Dlo[m * LDW + n0 + nl] = l; }
      }
    }
  };

  loadW(0, Wq);
  __syncthreads();
  project(sQ[0], sQ[1], 0);
  __syncthreads();
  loadW(1, Wk);
  __syncthreads();
  project(sK[0], sK[1], 0);
  __syncthreads();
  loadW(2, Wv);
  __syncthreads();
  project(sVt[0], sVt[1], 1);
  __syncthreads();

  // ---- logits = Q_h @ K_h^T / sqrt(d) ; wave w -> head w/2, rows 16*(w&1).. ----
  {
    const int h  = wave >> 1;
    const int m0 = (wave & 1) * 16;
    v16bf qh = loadA(sQ[0], LDW, m0, 32 * h, lane);
    v16bf ql = loadA(sQ[1], LDW, m0, 32 * h, lane);
    #pragma unroll
    for (int nt = 0; nt < 2; ++nt) {
      // B[k=d][n=c] = K[c][d]: sK row-major [c][d] is exactly the B layout
      v16bf kh = loadB(sK[0], LDW, nt * 16, 32 * h, lane);
      v16bf kl = loadB(sK[1], LDW, nt * 16, 32 * h, lane);
      v8f acc = {0.f,0.f,0.f,0.f,0.f,0.f,0.f,0.f};
      acc = mm3(qh, ql, kh, kl, acc);
      #pragma unroll
      for (int r = 0; r < 8; ++r) {
        int a = m0 + r + 8 * hh;
        int c = nt * 16 + nl;
        sLog[h * 1024 + a * 32 + c] = acc[r] * INV_SQRT_D;
      }
    }
  }
  __syncthreads();

  // ---- masked softmax, one row per thread (4 heads x 32 rows = 128 threads) ----
  if (tid < 128) {
    const int h = tid >> 5, a = tid & 31;
    const float* row = sLog + h * 1024 + a * 32;
    const int ma = sMask[a];
    float x[32];
    float mx = NEG_INF;
    #pragma unroll
    for (int c = 0; c < 32; ++c) {
      float v = (ma && sMask[c]) ? row[c] : NEG_INF;
      x[c] = v;
      mx = fmaxf(mx, v);
    }
    float s = 0.f;
    #pragma unroll
    for (int c = 0; c < 32; ++c) { float e = expf(x[c] - mx); x[c] = e; s += e; }
    float inv = 1.0f / s;
    #pragma unroll
    for (int c = 0; c < 32; ++c) {
      float p = (ma && sMask[c]) ? x[c] * inv : 0.f;   // attend * pair_mask
      unsigned short hb = f2bf(p);
      sPhi[a * LDW + h * 32 + c] = hb;
      sPlo[a * LDW + h * 32 + c] = f2bf(p - bf2f(hb));
    }
  }
  __syncthreads();

  // ---- O_h = P_h @ V_h ; store into sS (src tile is dead) ----
  {
    const int h  = wave >> 1;
    const int m0 = (wave & 1) * 16;
    v16bf ph = loadA(sPhi, LDW, m0, 32 * h, lane);
    v16bf pl = loadA(sPlo, LDW, m0, 32 * h, lane);
    #pragma unroll
    for (int nt = 0; nt < 2; ++nt) {
      // B[k=c][n=d] = V[c][d]: sVt stored [d][c] gives contiguous-k per lane
      v16bf vh = loadB(sVt[0], LDV, h * 32 + nt * 16, 0, lane);
      v16bf vl = loadB(sVt[1], LDV, h * 32 + nt * 16, 0, lane);
      v8f acc = {0.f,0.f,0.f,0.f,0.f,0.f,0.f,0.f};
      acc = mm3(ph, pl, vh, vl, acc);
      #pragma unroll
      for (int r = 0; r < 8; ++r) {
        int m = m0 + r + 8 * hh;
        int n = h * 32 + nt * 16 + nl;
        float v = acc[r];
        unsigned short hb = f2bf(v);
        sS[0][m * LDW + n] = hb;
        sS[1][m * LDW + n] = f2bf(v - bf2f(hb));
      }
    }
  }
  __syncthreads();
  loadW(3, Wo);          // overwrites sLog / sP (both dead now)
  __syncthreads();

  // ---- result = O @ W_O ----
  {
    const int n0 = wave * 16;
    v8f acc0 = {0.f,0.f,0.f,0.f,0.f,0.f,0.f,0.f};
    v8f acc1 = {0.f,0.f,0.f,0.f,0.f,0.f,0.f,0.f};
    #pragma unroll
    for (int kb = 0; kb < 4; ++kb) {
      int k0 = kb * 32;
      v16bf bh = loadB(sW[0], LDW, n0, k0, lane);
      v16bf bl = loadB(sW[1], LDW, n0, k0, lane);
      v16bf ah = loadA(sS[0], LDW, 0, k0, lane);
      v16bf al = loadA(sS[1], LDW, 0, k0, lane);
      acc0 = mm3(ah, al, bh, bl, acc0);
      ah = loadA(sS[0], LDW, 16, k0, lane);
      al = loadA(sS[1], LDW, 16, k0, lane);
      acc1 = mm3(ah, al, bh, bl, acc1);
    }
#if HAVE_TDM
    if (USE_WS) {
      // stage fp32 tile in LDS (sQ is dead) and DMA it out contiguously
      float* sOut = (float*)&sQ[0][0];          // 32*128 f32 = 16KB (< 17408B)
      #pragma unroll
      for (int r = 0; r < 8; ++r) {
        sOut[(r + 8 * hh) * 128 + n0 + nl]      = acc0[r];
        sOut[(16 + r + 8 * hh) * 128 + n0 + nl] = acc1[r];
      }
      __syncthreads();
      if (wave == 0) {
        tdm_store_1d(lds_offset(sOut), (unsigned long long)(uintptr_t)Og,
                     OUT_TILE_BYTES / 8);
        // S_ENDPGM performs an implicit wait-idle; TENSORcnt drains before exit.
      }
      return;
    }
#endif
    #pragma unroll
    for (int r = 0; r < 8; ++r) {
      Og[(r + 8 * hh) * 128 + n0 + nl]      = acc0[r];
      Og[(16 + r + 8 * hh) * 128 + n0 + nl] = acc1[r];
    }
  }
}

extern "C" void kernel_launch(void* const* d_in, const int* in_sizes, int n_in,
                              void* d_out, int out_size, void* d_ws, size_t ws_size,
                              hipStream_t stream) {
  const float* src  = (const float*)d_in[0];
  const int*   mask = (const int*)d_in[1];
  const float* Wq   = (const float*)d_in[2];
  const float* Wk   = (const float*)d_in[3];
  const float* Wv   = (const float*)d_in[4];
  const float* Wo   = (const float*)d_in[5];
  float* out = (float*)d_out;
  const int nTiles = in_sizes[0] / (32 * 128);   // B*T = 8192

  const size_t ws_needed = 4ull * W_BYTES;
  if (ws_size >= ws_needed) {
    unsigned short* wsbf = (unsigned short*)d_ws;
    prep_weights<<<dim3(4), dim3(256), 0, stream>>>(Wq, Wk, Wv, Wo, wsbf);
    term_attn_kernel<1><<<dim3(nTiles), dim3(256), 0, stream>>>(
        src, mask, Wq, Wk, Wv, Wo, wsbf, out);
  } else {
    term_attn_kernel<0><<<dim3(nTiles), dim3(256), 0, stream>>>(
        src, mask, Wq, Wk, Wv, Wo, nullptr, out);
  }
}
